// TransformerEncoderLayer_82944408420464
// MI455X (gfx1250) — compile-verified, matrix operand order checked by reading, not run on hardware
//
#include <hip/hip_runtime.h>
#include <hip/hip_bf16.h>

#define USE_ASYNC 1

typedef __attribute__((ext_vector_type(16))) __bf16 v16bf;
typedef __attribute__((ext_vector_type(16))) unsigned short v16u;
typedef __attribute__((ext_vector_type(8)))  float v8f;

union FragU { v16u u; v16bf b; };

#define Bd 4
#define Sd 1024
#define Ed 1024
#define Hd 16
#define Dd 64
#define FFd 4096

__device__ __forceinline__ unsigned short f2bf(float f) {
    unsigned int u = __float_as_uint(f);
    u += 0x7FFFu + ((u >> 16) & 1u);
    return (unsigned short)(u >> 16);
}

// A/B fragment, row-contiguous in contraction dim (two 16B ds_load_b128).
// Lane layout per CDNA5 ISA: lanes 0-15: K {0..7,16..23}; lanes 16-31: K {8..15,24..31}.
__device__ __forceinline__ v16bf ld_frag_row(const unsigned short* p, int hi) {
    FragU f;
    const unsigned short* q = p + hi * 8;
#pragma unroll
    for (int i = 0; i < 8; ++i) f.u[i] = q[i];
#pragma unroll
    for (int i = 0; i < 8; ++i) f.u[8 + i] = q[16 + i];
    return f.b;
}

// A fragment from f32 global memory (gnn_adj), converted to bf16.
__device__ __forceinline__ v16bf ld_frag_row_f32(const float* p, int hi) {
    FragU f;
    const float* q = p + hi * 8;
#pragma unroll
    for (int i = 0; i < 8; ++i) f.u[i] = f2bf(q[i]);
#pragma unroll
    for (int i = 0; i < 8; ++i) f.u[8 + i] = f2bf(q[16 + i]);
    return f.b;
}

__device__ __forceinline__ v8f wmma_bf16(v16bf a, v16bf b, v8f c) {
    return __builtin_amdgcn_wmma_f32_16x16x32_bf16(false, a, false, b,
                                                   (short)0, c, false, false);
}

// 16B global -> LDS copy; async (ASYNCcnt) on CDNA5.
__device__ __forceinline__ void copy16_g2l(const unsigned short* g, unsigned short* l) {
#if USE_ASYNC
    unsigned long long ga = (unsigned long long)g;
    unsigned lo = (unsigned)(unsigned long long)l;  // generic addr[31:0] == LDS offset
    asm volatile("global_load_async_to_lds_b128 %0, %1, off"
                 :: "v"(lo), "v"(ga) : "memory");
#else
    *(uint4*)l = *(const uint4*)g;
#endif
}

__device__ __forceinline__ void async_wait() {
#if USE_ASYNC
    asm volatile("s_wait_asynccnt 0x0" ::: "memory");
#endif
}

// ---------------- fp32 -> bf16 weight conversion ----------------
__global__ __launch_bounds__(256) void cvt_kernel(const float* __restrict__ src,
                                                  unsigned short* __restrict__ dst,
                                                  int n) {
    for (int i = blockIdx.x * 256 + threadIdx.x; i < n; i += gridDim.x * 256)
        dst[i] = f2bf(src[i]);
}

// ---------------- LayerNorm (one block per row of 1024) ----------------
__global__ __launch_bounds__(256) void ln_kernel(const float* __restrict__ x,
                                                 const float* __restrict__ g,
                                                 const float* __restrict__ be,
                                                 unsigned short* __restrict__ out) {
    __shared__ float red[256];
    int t = threadIdx.x;
    size_t row = blockIdx.x;
    const float* p = x + row * Ed;
    float v[4];
#pragma unroll
    for (int i = 0; i < 4; ++i) v[i] = p[t + i * 256];
    float s = v[0] + v[1] + v[2] + v[3];
    red[t] = s;
    __syncthreads();
    for (int o = 128; o > 0; o >>= 1) {
        if (t < o) red[t] += red[t + o];
        __syncthreads();
    }
    float mean = red[0] * (1.0f / Ed);
    __syncthreads();
    float s2 = 0.f;
#pragma unroll
    for (int i = 0; i < 4; ++i) { float d = v[i] - mean; s2 += d * d; }
    red[t] = s2;
    __syncthreads();
    for (int o = 128; o > 0; o >>= 1) {
        if (t < o) red[t] += red[t + o];
        __syncthreads();
    }
    float rstd = rsqrtf(red[0] * (1.0f / Ed) + 1e-5f);
    unsigned short* op = out + row * Ed;
#pragma unroll
    for (int i = 0; i < 4; ++i) {
        int idx = t + i * 256;
        op[idx] = f2bf((v[i] - mean) * rstd * g[idx] + be[idx]);
    }
}

// ---------------- generic bf16 WMMA GEMM, double-buffered pipeline ----------
// C[M,N] = A[M,K]@B[K,N] + bias;  MODE 0: *scale -> bf16
//                                 MODE 1: + res -> f32
//                                 MODE 2: gelu  -> bf16
// A staged row-major via async-to-LDS; B staged TRANSPOSED (n-major, k-contig)
// so all WMMA fragment loads are pairs of ds_load_b128. Next tile's async
// copies + global loads are issued before the current tile's WMMAs; the
// B transpose-scatter runs after the WMMAs so global latency hides under math.
template <int MODE>
__global__ __launch_bounds__(256) void gemm_kernel(
    const unsigned short* __restrict__ A, int lda,
    const unsigned short* __restrict__ B, int ldb,
    const float* __restrict__ bias, float scale,
    const float* __restrict__ res,
    float* __restrict__ outF, unsigned short* __restrict__ outB,
    int ldc, int K) {
    __shared__ unsigned short As[2][128][40];   // 80B rows, 16B aligned
    __shared__ unsigned short Bt[2][128][40];   // transposed: [n][k]

    int t = threadIdx.x;
    int wave = t >> 5, lane = t & 31;
    int wm = wave >> 2, wn = wave & 3;
    int hi = lane >> 4, lc = lane & 15;
    size_t rowBase = (size_t)blockIdx.y * 128;
    size_t colBase = (size_t)blockIdx.x * 128;

    v8f acc[4][2];
#pragma unroll
    for (int i = 0; i < 4; ++i)
#pragma unroll
        for (int j = 0; j < 2; ++j) acc[i][j] = (v8f){};

    int arow = t >> 1, akh = (t & 1) * 16;
    int bkp = (t >> 4) * 2;        // k-pair base 0..30
    int bcg = (t & 15) * 8;        // col group 0..120

    const unsigned short* aRow = A + (rowBase + arow) * lda + akh;
    const unsigned short* bRow0 = B + (size_t)bkp * ldb + colBase + bcg;
    const unsigned short* bRow1 = bRow0 + ldb;
    int nk = K >> 5;

    // prologue: stage tile 0 into buffer 0
    copy16_g2l(aRow, &As[0][arow][akh]);
    copy16_g2l(aRow + 8, &As[0][arow][akh + 8]);
    {
        uint4 r0 = *(const uint4*)bRow0;
        uint4 r1 = *(const uint4*)bRow1;
        const unsigned short* e0 = (const unsigned short*)&r0;
        const unsigned short* e1 = (const unsigned short*)&r1;
#pragma unroll
        for (int i = 0; i < 8; ++i)
            *(unsigned int*)&Bt[0][bcg + i][bkp] =
                (unsigned int)e0[i] | ((unsigned int)e1[i] << 16);
    }

    for (int kt = 0; kt < nk; ++kt) {
        int cur = kt & 1;
        async_wait();
        __syncthreads();

        bool pf = (kt + 1 < nk);
        uint4 r0 = {}, r1 = {};
        if (pf) {
            int k1 = (kt + 1) * 32;
            copy16_g2l(aRow + k1, &As[cur ^ 1][arow][akh]);
            copy16_g2l(aRow + k1 + 8, &As[cur ^ 1][arow][akh + 8]);
            r0 = *(const uint4*)(bRow0 + (size_t)k1 * ldb);
            r1 = *(const uint4*)(bRow1 + (size_t)k1 * ldb);
        }

        v16bf aF[4], bF[2];
#pragma unroll
        for (int mc = 0; mc < 4; ++mc)
            aF[mc] = ld_frag_row(&As[cur][wm * 64 + mc * 16 + lc][0], hi);
#pragma unroll
        for (int nc = 0; nc < 2; ++nc)
            bF[nc] = ld_frag_row(&Bt[cur][wn * 32 + nc * 16 + lc][0], hi);
#pragma unroll
        for (int mc = 0; mc < 4; ++mc)
#pragma unroll
            for (int nc = 0; nc < 2; ++nc)
                acc[mc][nc] = wmma_bf16(aF[mc], bF[nc], acc[mc][nc]);

        if (pf) {
            const unsigned short* e0 = (const unsigned short*)&r0;
            const unsigned short* e1 = (const unsigned short*)&r1;
#pragma unroll
            for (int i = 0; i < 8; ++i)
                *(unsigned int*)&Bt[cur ^ 1][bcg + i][bkp] =
                    (unsigned int)e0[i] | ((unsigned int)e1[i] << 16);
        }
    }

#pragma unroll
    for (int mc = 0; mc < 4; ++mc)
#pragma unroll
        for (int nc = 0; nc < 2; ++nc) {
            size_t col = colBase + wn * 32 + nc * 16 + lc;
            float bv = bias[col];
#pragma unroll
            for (int j = 0; j < 8; ++j) {
                size_t row = rowBase + wm * 64 + mc * 16 + j + 8 * hi;
                float x = acc[mc][nc][j] + bv;
                size_t idx = row * (size_t)ldc + col;
                if (MODE == 0) {
                    outB[idx] = f2bf(x * scale);
                } else if (MODE == 1) {
                    outF[idx] = x + res[idx];
                } else {
                    float ge = 0.5f * x * (1.0f + erff(x * 0.70710678118654752f));
                    outB[idx] = f2bf(ge);
                }
            }
        }
}

// ------- flash attention + post-softmax gnn term, double-buffered ---------
__global__ __launch_bounds__(256) void attn_kernel(
    const unsigned short* __restrict__ q, const unsigned short* __restrict__ k,
    const unsigned short* __restrict__ v, const float* __restrict__ mask,
    const float* __restrict__ gnn, unsigned short* __restrict__ xv) {
    __shared__ unsigned short Ks[2][32][64];    // [key][d]
    __shared__ unsigned short Vt[2][64][40];    // transposed: [d][key]
    __shared__ unsigned short Ps[8][16][32];    // wave-private P tiles

    int t = threadIdx.x;
    int wave = t >> 5, lane = t & 31;
    int hi = lane >> 4, lc = lane & 15;
    int qblk = blockIdx.x & 7;
    int bh = blockIdx.x >> 3;
    int b = bh >> 4, h = bh & 15;
    int qbase = qblk * 128 + wave * 16;

    const unsigned short* qp = q + (size_t)b * Sd * Ed + h * Dd;
    const unsigned short* kp = k + (size_t)b * Sd * Ed + h * Dd;
    const unsigned short* vp = v + (size_t)b * Sd * Ed + h * Dd;
    const float* mp = mask + (size_t)b * Sd * Sd;
    const float* gp = gnn + (size_t)b * Sd * Sd;

    // preload Q fragments (16 rows x 64 d, two K-chunks of 32)
    v16bf qf0 = ld_frag_row(qp + (size_t)(qbase + lc) * Ed, hi);
    v16bf qf1 = ld_frag_row(qp + (size_t)(qbase + lc) * Ed + 32, hi);

    float mrow[8], lrow[8];
    v8f accP[4], accG[4];
#pragma unroll
    for (int j = 0; j < 8; ++j) { mrow[j] = -1e30f; lrow[j] = 0.f; }
#pragma unroll
    for (int dc = 0; dc < 4; ++dc) { accP[dc] = (v8f){}; accG[dc] = (v8f){}; }

    int klr = t >> 3, klc = (t & 7) * 8;   // K staging: one b128/thread
    int vkp = (t >> 4) * 2;                // V staging: key-pair 0..30
    int vdg = (t & 15) * 4;                // d group of 4: 0..60

    const unsigned short* kRow = kp + (size_t)klr * Ed + klc;
    const unsigned short* vRow0 = vp + (size_t)vkp * Ed + vdg;
    const unsigned short* vRow1 = vRow0 + Ed;
    const int NK = Sd / 32;

    // prologue: stage tile 0 into buffer 0
    copy16_g2l(kRow, &Ks[0][klr][klc]);
    {
        uint2 r0 = *(const uint2*)vRow0;
        uint2 r1 = *(const uint2*)vRow1;
        const unsigned short* e0 = (const unsigned short*)&r0;
        const unsigned short* e1 = (const unsigned short*)&r1;
#pragma unroll
        for (int i = 0; i < 4; ++i)
            *(unsigned int*)&Vt[0][vdg + i][vkp] =
                (unsigned int)e0[i] | ((unsigned int)e1[i] << 16);
    }

    for (int kt = 0; kt < NK; ++kt) {
        int cur = kt & 1;
        int kg0 = kt * 32;
        async_wait();
        __syncthreads();

        bool pf = (kt + 1 < NK);
        uint2 r0 = {}, r1 = {};
        if (pf) {
            size_t koff = (size_t)(kt + 1) * 32 * Ed;
            copy16_g2l(kRow + koff, &Ks[cur ^ 1][klr][klc]);
            r0 = *(const uint2*)(vRow0 + koff);
            r1 = *(const uint2*)(vRow1 + koff);
        }

        // scores for 32 keys: S = Q(16x64) @ K^T
        v8f s0 = (v8f){}, s1 = (v8f){};
        s0 = wmma_bf16(qf0, ld_frag_row(&Ks[cur][lc][0], hi), s0);
        s0 = wmma_bf16(qf1, ld_frag_row(&Ks[cur][lc][32], hi), s0);
        s1 = wmma_bf16(qf0, ld_frag_row(&Ks[cur][16 + lc][0], hi), s1);
        s1 = wmma_bf16(qf1, ld_frag_row(&Ks[cur][16 + lc][32], hi), s1);

        // mask + online softmax (all-masked-chunk garbage flushed by
        // corr = exp(-1e30 - m_real) == 0 once a real key appears)
#pragma unroll
        for (int j = 0; j < 8; ++j) {
            int qi = qbase + j + 8 * hi;
            float mv0 = mp[(size_t)qi * Sd + kg0 + lc];
            float mv1 = mp[(size_t)qi * Sd + kg0 + 16 + lc];
            float a0 = (mv0 == 0.f) ? -1e30f : s0[j];
            float a1 = (mv1 == 0.f) ? -1e30f : s1[j];
            float mx = fmaxf(a0, a1);
            mx = fmaxf(mx, __shfl_xor(mx, 1, 32));
            mx = fmaxf(mx, __shfl_xor(mx, 2, 32));
            mx = fmaxf(mx, __shfl_xor(mx, 4, 32));
            mx = fmaxf(mx, __shfl_xor(mx, 8, 32));
            float mnew = fmaxf(mrow[j], mx);
            float corr = __expf(mrow[j] - mnew);
            float p0 = __expf(a0 - mnew);
            float p1 = __expf(a1 - mnew);
            float rs = p0 + p1;
            rs += __shfl_xor(rs, 1, 32);
            rs += __shfl_xor(rs, 2, 32);
            rs += __shfl_xor(rs, 4, 32);
            rs += __shfl_xor(rs, 8, 32);
            lrow[j] = lrow[j] * corr + rs;
            mrow[j] = mnew;
#pragma unroll
            for (int dc = 0; dc < 4; ++dc) accP[dc][j] *= corr;
            int m = j + 8 * hi;
            Ps[wave][m][lc] = f2bf(p0);
            Ps[wave][m][16 + lc] = f2bf(p1);
        }

        // P (wave-private LDS round-trip) and G (direct global) A-fragments
        v16bf pfg = ld_frag_row(&Ps[wave][lc][0], hi);
        v16bf gf = ld_frag_row_f32(gp + (size_t)(qbase + lc) * Sd + kg0, hi);
#pragma unroll
        for (int dc = 0; dc < 4; ++dc) {
            v16bf vf = ld_frag_row(&Vt[cur][dc * 16 + lc][0], hi);
            accP[dc] = wmma_bf16(pfg, vf, accP[dc]);
            accG[dc] = wmma_bf16(gf, vf, accG[dc]);
        }

        if (pf) {
            const unsigned short* e0 = (const unsigned short*)&r0;
            const unsigned short* e1 = (const unsigned short*)&r1;
#pragma unroll
            for (int i = 0; i < 4; ++i)
                *(unsigned int*)&Vt[cur ^ 1][vdg + i][vkp] =
                    (unsigned int)e0[i] | ((unsigned int)e1[i] << 16);
        }
    }

#pragma unroll
    for (int dc = 0; dc < 4; ++dc)
#pragma unroll
        for (int j = 0; j < 8; ++j) {
            int qi = qbase + j + 8 * hi;
            float o = accP[dc][j] / lrow[j] + accG[dc][j];
            xv[(size_t)(b * Sd + qi) * Ed + h * Dd + dc * 16 + lc] = f2bf(o);
        }
}

// ---------------- host-side orchestration ----------------
extern "C" void kernel_launch(void* const* d_in, const int* in_sizes, int n_in,
                              void* d_out, int out_size, void* d_ws, size_t ws_size,
                              hipStream_t stream) {
    const float* x    = (const float*)d_in[0];
    const float* mask = (const float*)d_in[1];
    const float* gnn  = (const float*)d_in[2];
    const float* Wq   = (const float*)d_in[3];
    const float* bq   = (const float*)d_in[4];
    const float* Wk   = (const float*)d_in[5];
    const float* bk   = (const float*)d_in[6];
    const float* Wv   = (const float*)d_in[7];
    const float* bv   = (const float*)d_in[8];
    const float* Wo   = (const float*)d_in[9];
    const float* bo   = (const float*)d_in[10];
    const float* g1   = (const float*)d_in[11];
    const float* b1   = (const float*)d_in[12];
    const float* g2   = (const float*)d_in[13];
    const float* b2   = (const float*)d_in[14];
    const float* W1   = (const float*)d_in[15];
    const float* bf1  = (const float*)d_in[16];
    const float* W2   = (const float*)d_in[17];
    const float* bf2  = (const float*)d_in[18];
    float* out = (float*)d_out;

    char* w = (char*)d_ws;
    const size_t MB = 1024 * 1024;
    unsigned short* Wq_b = (unsigned short*)(w + 0 * MB);
    unsigned short* Wk_b = (unsigned short*)(w + 2 * MB);
    unsigned short* Wv_b = (unsigned short*)(w + 4 * MB);
    unsigned short* Wo_b = (unsigned short*)(w + 6 * MB);
    unsigned short* W1_b = (unsigned short*)(w + 8 * MB);
    unsigned short* W2_b = (unsigned short*)(w + 16 * MB);
    unsigned short* qb   = (unsigned short*)(w + 24 * MB);
    unsigned short* kb   = (unsigned short*)(w + 32 * MB);
    unsigned short* vb   = (unsigned short*)(w + 40 * MB);
    unsigned short* bufA = (unsigned short*)(w + 48 * MB);  // nx -> xv -> h
    unsigned short* ffb  = (unsigned short*)(w + 56 * MB);  // 32 MB

    const int M = Bd * Sd;  // 4096

    // 1) weights -> bf16
    cvt_kernel<<<2048, 256, 0, stream>>>(Wq, Wq_b, Ed * Ed);
    cvt_kernel<<<2048, 256, 0, stream>>>(Wk, Wk_b, Ed * Ed);
    cvt_kernel<<<2048, 256, 0, stream>>>(Wv, Wv_b, Ed * Ed);
    cvt_kernel<<<2048, 256, 0, stream>>>(Wo, Wo_b, Ed * Ed);
    cvt_kernel<<<4096, 256, 0, stream>>>(W1, W1_b, Ed * FFd);
    cvt_kernel<<<4096, 256, 0, stream>>>(W2, W2_b, FFd * Ed);

    // 2) LN1 -> nx (bufA)
    ln_kernel<<<M, 256, 0, stream>>>(x, g1, b1, bufA);

    // 3) QKV projections (1/sqrt(D) folded into Q)
    dim3 blk(256);
    dim3 gE(Ed / 128, M / 128);
    gemm_kernel<0><<<gE, blk, 0, stream>>>(bufA, Ed, Wq_b, Ed, bq, 0.125f,
                                           nullptr, nullptr, qb, Ed, Ed);
    gemm_kernel<0><<<gE, blk, 0, stream>>>(bufA, Ed, Wk_b, Ed, bk, 1.0f,
                                           nullptr, nullptr, kb, Ed, Ed);
    gemm_kernel<0><<<gE, blk, 0, stream>>>(bufA, Ed, Wv_b, Ed, bv, 1.0f,
                                           nullptr, nullptr, vb, Ed, Ed);

    // 4) attention -> xv (bufA)
    attn_kernel<<<Bd * Hd * (Sd / 128), blk, 0, stream>>>(qb, kb, vb, mask, gnn, bufA);

    // 5) O projection + bias + residual(x) -> fp32 in d_out
    gemm_kernel<1><<<gE, blk, 0, stream>>>(bufA, Ed, Wo_b, Ed, bo, 1.0f,
                                           x, out, nullptr, Ed, Ed);

    // 6) LN2 -> h (bufA)
    ln_kernel<<<M, 256, 0, stream>>>(out, g2, b2, bufA);

    // 7) FFN1 + GeLU -> ff
    dim3 gF1(FFd / 128, M / 128);
    gemm_kernel<2><<<gF1, blk, 0, stream>>>(bufA, Ed, W1_b, FFd, bf1, 1.0f,
                                            nullptr, nullptr, ffb, FFd, Ed);

    // 8) FFN2 + bias + residual (in-place on d_out)
    gemm_kernel<1><<<gE, blk, 0, stream>>>(ffb, FFd, W2_b, Ed, bf2, 1.0f,
                                           out, out, nullptr, Ed, FFd);

    (void)in_sizes; (void)n_in; (void)out_size; (void)ws_size;
}
